// Attention_26998164423112
// MI455X (gfx1250) — compile-verified, matrix operand order checked by reading, not run on hardware
//
#include <hip/hip_runtime.h>
#include <math.h>

// ---------------------------------------------------------------------------
// Problem constants (from reference): B=4, Q=K=1024, C=256, H=8, D=32
// ---------------------------------------------------------------------------
#define BATCH 4
#define SEQ   1024
#define CDIM  256
#define HEADS 8
#define HDIM  32
#define MTOK  (BATCH * SEQ)   // 4096 rows for all token-major GEMMs

typedef __attribute__((ext_vector_type(16))) __bf16 v16bf;
typedef __attribute__((ext_vector_type(8)))  float  v8f;
typedef __attribute__((ext_vector_type(2)))  __bf16 bf16x2;
typedef int v4i __attribute__((vector_size(16)));          // matches builtin sig
typedef v4i __attribute__((address_space(1)))* g_v4i_p;    // global int4*
typedef v4i __attribute__((address_space(3)))* l_v4i_p;    // LDS int4*

union FragA { v16bf v; bf16x2 p[8]; };
union FragC { v8f  v; float  f[8]; };

// K-pair base index held by VGPR v for 16-bit A/B fragments (ISA 7.12.2):
// lanes 0-15: V0..V3 -> K={0,2,4,6}, V4..V7 -> K={16,18,20,22}; lanes 16-31: +8
__device__ __forceinline__ int kpair(int v, int half) {
  return ((v < 4) ? (2 * v) : (2 * v + 8)) + 8 * half;
}

// Load a 16x32 bf16 fragment whose contraction dim is contiguous in memory.
// 'row' is the per-lane M (A-matrix) or N (B-matrix) line index.
__device__ __forceinline__ v16bf load_frag(const __bf16* base, int ld, int row,
                                           int k0, int half) {
  FragA f;
#pragma unroll
  for (int v = 0; v < 8; ++v)
    f.p[v] = *(const bf16x2*)(base + (size_t)row * ld + k0 + kpair(v, half));
  return f.v;
}

// 16-byte global -> LDS copy. Prefer the CDNA5 async path (ASYNCcnt).
#if defined(__has_builtin)
#if __has_builtin(__builtin_amdgcn_global_load_async_to_lds_b128)
#define HAVE_ASYNC_LDS 1
#endif
#endif

__device__ __forceinline__ void copy16_g2l(const __bf16* g, __bf16* l) {
#ifdef HAVE_ASYNC_LDS
  v4i* gp = (v4i*)g;   // drop const / retype within generic AS
  v4i* lp = (v4i*)l;
  __builtin_amdgcn_global_load_async_to_lds_b128((g_v4i_p)gp, (l_v4i_p)lp, 0, 0);
#else
  *(uint4*)l = *(const uint4*)g;
#endif
}

__device__ __forceinline__ void wait_async_lds() {
#ifdef HAVE_ASYNC_LDS
  asm volatile("s_wait_asynccnt 0" ::: "memory");
#else
  asm volatile("s_wait_dscnt 0" ::: "memory");
#endif
}

// ---------------------------------------------------------------------------
// f32 -> bf16 conversion
// ---------------------------------------------------------------------------
__global__ __launch_bounds__(256) void cvt_bf16(const float* __restrict__ src,
                                                __bf16* __restrict__ dst, int n) {
  int i = blockIdx.x * 256 + threadIdx.x;
  if (i < n) dst[i] = (__bf16)src[i];
}

// ---------------------------------------------------------------------------
// Projection GEMM: Y[4096,256] = X[4096,256] @ W[256,256]^T  (bf16 WMMA)
// One wave computes a 16x64 output strip (A fragment reused 4x).
// MODE 0: q -> head layout [B,H,Q,D] bf16, scaled by 1/sqrt(D)
// MODE 1: k -> head layout [B,H,K,D] bf16
// MODE 2: v -> transposed  [B,H,D,K] bf16 (so PV contraction is contiguous)
// MODE 3: gate -> sigmoid(y + bg) f32 [B*Q, H*D]
// ---------------------------------------------------------------------------
template <int MODE>
__global__ __launch_bounds__(256) void gemm_proj(const __bf16* __restrict__ X,
                                                 const __bf16* __restrict__ W,
                                                 const float* __restrict__ bias,
                                                 void* __restrict__ outv) {
  const int wid  = blockIdx.x * 8 + (threadIdx.x >> 5);  // 0..1023
  const int tm   = wid >> 2;            // 0..255 (M tiles of 16)
  const int tng  = wid & 3;             // 0..3   (N groups of 64)
  const int lane = threadIdx.x & 31;
  const int half = lane >> 4;
  const int ln   = lane & 15;

  const v8f z = {0.f, 0.f, 0.f, 0.f, 0.f, 0.f, 0.f, 0.f};
  FragC acc[4];
#pragma unroll
  for (int j = 0; j < 4; ++j) acc[j].v = z;

#pragma unroll 2
  for (int c0 = 0; c0 < CDIM; c0 += 32) {
    const v16bf a = load_frag(X, CDIM, tm * 16 + ln, c0, half);
#pragma unroll
    for (int j = 0; j < 4; ++j) {
      const v16bf b = load_frag(W, CDIM, tng * 64 + j * 16 + ln, c0, half);
      acc[j].v = __builtin_amdgcn_wmma_f32_16x16x32_bf16(false, a, false, b,
                                                         (short)0, acc[j].v,
                                                         false, false);
    }
  }

#pragma unroll
  for (int j = 0; j < 4; ++j) {
#pragma unroll
    for (int v = 0; v < 8; ++v) {
      const int m = tm * 16 + v + 8 * half;       // token row
      const int n = tng * 64 + j * 16 + ln;       // output channel
      const float y = acc[j].f[v];
      const int b_ = m >> 10, qi = m & 1023, h = n >> 5, d = n & 31;
      if (MODE == 0) {
        ((__bf16*)outv)[(((size_t)b_ * HEADS + h) * SEQ + qi) * HDIM + d] =
            (__bf16)(y * 0.17677669529663687f);   // 1/sqrt(32)
      } else if (MODE == 1) {
        ((__bf16*)outv)[(((size_t)b_ * HEADS + h) * SEQ + qi) * HDIM + d] = (__bf16)y;
      } else if (MODE == 2) {
        ((__bf16*)outv)[(((size_t)b_ * HEADS + h) * HDIM + d) * SEQ + qi] = (__bf16)y;
      } else {
        const float t = y + bias[n];
        ((float*)outv)[(size_t)m * CDIM + n] = 1.f / (1.f + __expf(-t));
      }
    }
  }
}

// ---------------------------------------------------------------------------
// Output GEMM: out[4096,256] = A[4096,256] @ Wo[256,256]^T + bo  (f32 out)
// Same 16x64 wave strip.
// ---------------------------------------------------------------------------
__global__ __launch_bounds__(256) void gemm_out(const __bf16* __restrict__ X,
                                                const __bf16* __restrict__ W,
                                                const float* __restrict__ bias,
                                                float* __restrict__ out) {
  const int wid  = blockIdx.x * 8 + (threadIdx.x >> 5);
  const int tm   = wid >> 2;
  const int tng  = wid & 3;
  const int lane = threadIdx.x & 31;
  const int half = lane >> 4;
  const int ln   = lane & 15;

  const v8f z = {0.f, 0.f, 0.f, 0.f, 0.f, 0.f, 0.f, 0.f};
  FragC acc[4];
#pragma unroll
  for (int j = 0; j < 4; ++j) acc[j].v = z;

#pragma unroll 2
  for (int c0 = 0; c0 < CDIM; c0 += 32) {
    const v16bf a = load_frag(X, CDIM, tm * 16 + ln, c0, half);
#pragma unroll
    for (int j = 0; j < 4; ++j) {
      const v16bf b = load_frag(W, CDIM, tng * 64 + j * 16 + ln, c0, half);
      acc[j].v = __builtin_amdgcn_wmma_f32_16x16x32_bf16(false, a, false, b,
                                                         (short)0, acc[j].v,
                                                         false, false);
    }
  }
#pragma unroll
  for (int j = 0; j < 4; ++j) {
#pragma unroll
    for (int v = 0; v < 8; ++v) {
      const int m = tm * 16 + v + 8 * half;
      const int n = tng * 64 + j * 16 + ln;
      out[(size_t)m * CDIM + n] = acc[j].f[v] + bias[n];
    }
  }
}

// ---------------------------------------------------------------------------
// Flash attention.
// grid = (B*H, Q/128), block = 256 (8 waves). Each wave owns 16 query rows.
// K/V tiles (32 kv) are staged once per WORKGROUP into LDS via async copies
// (ASYNCcnt) and consumed by all 8 waves -> 8x less L2 traffic on K/V.
// ---------------------------------------------------------------------------
#define KVPAD 40   // 32 bf16 payload + 8 pad: keeps 16B alignment, spreads banks

__global__ __launch_bounds__(256) void attn(const __bf16* __restrict__ qh,
                                            const __bf16* __restrict__ kh,
                                            const __bf16* __restrict__ vT,
                                            const float* __restrict__ bmask,
                                            const float* __restrict__ bpair,
                                            const float* __restrict__ gbuf,
                                            __bf16* __restrict__ attout) {
  __shared__ __align__(16) __bf16 ktile[32 * KVPAD];  // [kv][d]
  __shared__ __align__(16) __bf16 vtile[32 * KVPAD];  // [d][kv]
  __shared__ __align__(16) __bf16 pbuf[8][16 * 34];   // per-wave P staging

  const int bh   = blockIdx.x;           // 0..31
  const int b    = bh >> 3, h = bh & 7;
  const int wave = threadIdx.x >> 5;
  const int lane = threadIdx.x & 31;
  const int half = lane >> 4;
  const int ln   = lane & 15;
  const int qbase = blockIdx.y * 128 + wave * 16;

  const __bf16* qp = qh + ((size_t)bh * SEQ + qbase) * HDIM;
  const __bf16* kp = kh + (size_t)bh * SEQ * HDIM;
  const __bf16* vp = vT + (size_t)bh * HDIM * SEQ;
  const float*  bm = bmask + (size_t)b * SEQ;
  const float*  bp = bpair + ((size_t)bh * SEQ + qbase) * SEQ;

  // cooperative copy assignment: thread -> one 16B chunk of K or V tile
  const int crow = (threadIdx.x >> 2) & 31;   // 0..31
  const int cch  = (threadIdx.x & 3) * 8;     // element offset 0,8,16,24
  const bool isV = threadIdx.x >= 128;

  const v16bf qa = load_frag(qp, HDIM, ln, 0, half);   // Q A-fragment (reused)

  const v8f z = {0.f, 0.f, 0.f, 0.f, 0.f, 0.f, 0.f, 0.f};
  FragC o0, o1; o0.v = z; o1.v = z;
  float mrow[8], lrow[8];
#pragma unroll
  for (int v = 0; v < 8; ++v) { mrow[v] = -3.0e38f; lrow[v] = 0.f; }

  __bf16* pb = &pbuf[wave][0];

  for (int kv0 = 0; kv0 < SEQ; kv0 += 32) {
    // ---- stage K[kv0..kv0+31][0..31] and V^T[d][kv0..kv0+31] into LDS ----
    if (!isV) {
      copy16_g2l(kp + (size_t)(kv0 + crow) * HDIM + cch, ktile + crow * KVPAD + cch);
    } else {
      copy16_g2l(vp + (size_t)crow * SEQ + kv0 + cch,    vtile + crow * KVPAD + cch);
    }
    wait_async_lds();
    __syncthreads();

    // ---- scores: S[16 x 32] = Q[16x32] x K^T, contraction over D=32 ----
    const v16bf kb0 = load_frag(ktile, KVPAD, ln,      0, half);
    const v16bf kb1 = load_frag(ktile, KVPAD, 16 + ln, 0, half);
    FragC s0, s1;
    s0.v = __builtin_amdgcn_wmma_f32_16x16x32_bf16(false, qa, false, kb0,
                                                   (short)0, z, false, false);
    s1.v = __builtin_amdgcn_wmma_f32_16x16x32_bf16(false, qa, false, kb1,
                                                   (short)0, z, false, false);

    const float bm0 = bm[kv0 + ln];
    const float bm1 = bm[kv0 + 16 + ln];
    if (kv0 + 32 < SEQ)   // hint the bandwidth-critical bias_pair stream
      __builtin_prefetch(bp + (size_t)(8 * half) * SEQ + kv0 + 32 + ln, 0, 0);

    // ---- bias add + online softmax, per owned row (C-layout row v+8*half) --
#pragma unroll
    for (int v = 0; v < 8; ++v) {
      const int qr = v + 8 * half;
      float x0 = s0.f[v] + bm0 + bp[(size_t)qr * SEQ + kv0 + ln];
      float x1 = s1.f[v] + bm1 + bp[(size_t)qr * SEQ + kv0 + 16 + ln];
      float cmax = fmaxf(x0, x1);
#pragma unroll
      for (int msk = 1; msk < 16; msk <<= 1)
        cmax = fmaxf(cmax, __shfl_xor(cmax, msk));
      const float mnew  = fmaxf(mrow[v], cmax);
      const float scale = __expf(mrow[v] - mnew);
      const float p0 = __expf(x0 - mnew);
      const float p1 = __expf(x1 - mnew);
      float rs = p0 + p1;
#pragma unroll
      for (int msk = 1; msk < 16; msk <<= 1)
        rs += __shfl_xor(rs, msk);
      lrow[v] = lrow[v] * scale + rs;
      mrow[v] = mnew;
      o0.f[v] *= scale;
      o1.f[v] *= scale;
      pb[qr * 34 + ln]      = (__bf16)p0;   // C-layout -> LDS (A-layout transpose)
      pb[qr * 34 + 16 + ln] = (__bf16)p1;
    }
    asm volatile("s_wait_dscnt 0" ::: "memory");
    __builtin_amdgcn_wave_barrier();

    // ---- O += P[16x32] x V[32x32], contraction over kv tile of 32 ----
    const v16bf pa  = load_frag(pb, 34, ln, 0, half);
    const v16bf vb0 = load_frag(vtile, KVPAD, ln,      0, half);  // d 0..15
    const v16bf vb1 = load_frag(vtile, KVPAD, 16 + ln, 0, half);  // d 16..31
    o0.v = __builtin_amdgcn_wmma_f32_16x16x32_bf16(false, pa, false, vb0,
                                                   (short)0, o0.v, false, false);
    o1.v = __builtin_amdgcn_wmma_f32_16x16x32_bf16(false, pa, false, vb1,
                                                   (short)0, o1.v, false, false);

    __syncthreads();   // all waves done with the tiles before next overwrite
  }

  // ---- normalize, gate, store bf16 in [B, Q, H*D] for the output GEMM ----
#pragma unroll
  for (int v = 0; v < 8; ++v) {
    const int qi = qbase + v + 8 * half;
    const float inv = 1.f / lrow[v];
    const size_t gi = (((size_t)b * SEQ + qi) * CDIM) + h * HDIM;
    attout[gi + ln]      = (__bf16)(o0.f[v] * inv * gbuf[gi + ln]);
    attout[gi + 16 + ln] = (__bf16)(o1.f[v] * inv * gbuf[gi + 16 + ln]);
  }
}

// ---------------------------------------------------------------------------
// Launcher
// ---------------------------------------------------------------------------
extern "C" void kernel_launch(void* const* d_in, const int* in_sizes, int n_in,
                              void* d_out, int out_size, void* d_ws, size_t ws_size,
                              hipStream_t stream) {
  (void)in_sizes; (void)n_in; (void)out_size; (void)ws_size;
  const float* q_x   = (const float*)d_in[0];
  const float* kv_x  = (const float*)d_in[1];
  const float* bmask = (const float*)d_in[2];
  const float* bpair = (const float*)d_in[3];
  const float* Wq    = (const float*)d_in[4];
  const float* Wk    = (const float*)d_in[5];
  const float* Wv    = (const float*)d_in[6];
  const float* Wo    = (const float*)d_in[7];
  const float* bo    = (const float*)d_in[8];
  const float* Wg    = (const float*)d_in[9];
  const float* bg    = (const float*)d_in[10];

  char* ws = (char*)d_ws;
  size_t off = 0;
  auto alloc = [&](size_t bytes) {
    off = (off + 255) & ~(size_t)255;
    void* p = ws + off;
    off += bytes;
    return p;
  };

  __bf16* xq   = (__bf16*)alloc((size_t)MTOK * CDIM * 2);
  __bf16* xkv  = (__bf16*)alloc((size_t)MTOK * CDIM * 2);
  __bf16* wq   = (__bf16*)alloc((size_t)CDIM * CDIM * 2);
  __bf16* wk   = (__bf16*)alloc((size_t)CDIM * CDIM * 2);
  __bf16* wv   = (__bf16*)alloc((size_t)CDIM * CDIM * 2);
  __bf16* wo   = (__bf16*)alloc((size_t)CDIM * CDIM * 2);
  __bf16* wg   = (__bf16*)alloc((size_t)CDIM * CDIM * 2);
  __bf16* qhp  = (__bf16*)alloc((size_t)MTOK * CDIM * 2);   // [B,H,Q,D]
  __bf16* khp  = (__bf16*)alloc((size_t)MTOK * CDIM * 2);   // [B,H,K,D]
  __bf16* vtp  = (__bf16*)alloc((size_t)MTOK * CDIM * 2);   // [B,H,D,K]
  float*  gbuf = (float*) alloc((size_t)MTOK * CDIM * 4);   // sigmoid gate
  __bf16* aout = (__bf16*)alloc((size_t)MTOK * CDIM * 2);   // gated attn out

  const int nx = MTOK * CDIM;        // 1,048,576
  const int nw = CDIM * CDIM;        // 65,536
  cvt_bf16<<<(nx + 255) / 256, 256, 0, stream>>>(q_x,  xq,  nx);
  cvt_bf16<<<(nx + 255) / 256, 256, 0, stream>>>(kv_x, xkv, nx);
  cvt_bf16<<<(nw + 255) / 256, 256, 0, stream>>>(Wq, wq, nw);
  cvt_bf16<<<(nw + 255) / 256, 256, 0, stream>>>(Wk, wk, nw);
  cvt_bf16<<<(nw + 255) / 256, 256, 0, stream>>>(Wv, wv, nw);
  cvt_bf16<<<(nw + 255) / 256, 256, 0, stream>>>(Wo, wo, nw);
  cvt_bf16<<<(nw + 255) / 256, 256, 0, stream>>>(Wg, wg, nw);

  // (M/16)*(N/64) = 1024 wave-strips / 8 waves per block = 128 blocks
  gemm_proj<0><<<128, 256, 0, stream>>>(xq,  wq, nullptr, qhp);
  gemm_proj<1><<<128, 256, 0, stream>>>(xkv, wk, nullptr, khp);
  gemm_proj<2><<<128, 256, 0, stream>>>(xkv, wv, nullptr, vtp);
  gemm_proj<3><<<128, 256, 0, stream>>>(xq,  wg, bg,      gbuf);

  attn<<<dim3(BATCH * HEADS, SEQ / 128), 256, 0, stream>>>(
      qhp, khp, vtp, bmask, bpair, gbuf, aout);

  gemm_out<<<128, 256, 0, stream>>>(aout, wo, bo, (float*)d_out);
}
